// dynedgegauss_23029614641724
// MI455X (gfx1250) — compile-verified, matrix operand order checked by reading, not run on hardware
//
#include <hip/hip_runtime.h>
#include <hip/hip_bf16.h>

typedef __attribute__((ext_vector_type(16))) _Float16 v16h;
typedef __attribute__((ext_vector_type(8)))  _Float16 v8h;
typedef __attribute__((ext_vector_type(8)))  float    v8f;
typedef __attribute__((ext_vector_type(4)))  unsigned int u32x4;
typedef __attribute__((ext_vector_type(8)))  int      i32x8;
typedef __attribute__((ext_vector_type(4)))  int      i32x4;

#define AS1 __attribute__((address_space(1)))
#define AS3 __attribute__((address_space(3)))

#if defined(__has_builtin)
#if __has_builtin(__builtin_amdgcn_global_load_async_to_lds_b128) && \
    __has_builtin(__builtin_amdgcn_s_wait_asynccnt)
#define USE_ASYNC_LDS 1
#endif
#if __has_builtin(__builtin_amdgcn_tensor_load_to_lds) && \
    __has_builtin(__builtin_amdgcn_s_wait_tensorcnt)
#define USE_TDM 1
#endif
#endif
#ifndef USE_ASYNC_LDS
#define USE_ASYNC_LDS 0
#endif
#ifndef USE_TDM
#define USE_TDM 0
#endif

#define G_NODES 65536   // B*N = 128*512
#define NB      128
#define NN      512
#define KNBR    4
#define FOUT    192     // all edge-conv output widths

__device__ __forceinline__ float lrelu(float v) { return v > 0.0f ? v : 0.01f * v; }

// Load a 16x32 f16 WMMA fragment (A or B^T) from row-major LDS.
// Lane l (0-15): row = l, halves K[ko..ko+7] then K[ko+16..ko+23] with ko=0.
// Lane l (16-31): same row, ko=8. Matches CDNA5 16-bit A/B VGPR layout.
__device__ __forceinline__ v16h load_frag(const _Float16* base, int row, int rowStride, int ko) {
    const v8h lo = *(const v8h*)(base + row * rowStride + ko);
    const v8h hi = *(const v8h*)(base + row * rowStride + ko + 16);
    v16h r;
#pragma unroll
    for (int i = 0; i < 8; ++i) { r[i] = lo[i]; r[i + 8] = hi[i]; }
    return r;
}

// 16-byte global -> LDS copy: async (CDNA5 path, tracked by ASYNCcnt) or via VGPRs.
__device__ __forceinline__ void copy16_g2l(const _Float16* g, _Float16* l) {
#if USE_ASYNC_LDS
    __builtin_amdgcn_global_load_async_to_lds_b128((AS1 i32x4*)g, (AS3 i32x4*)l, 0, 0);
#else
    *(v8h*)l = *(const v8h*)g;
#endif
}

// ---------------- weight prep: f32 (K x Nn) -> f16 transposed padded (Np x Kp) ----------
__global__ void prep_weight(const float* __restrict__ W, const float* __restrict__ Wsub,
                            _Float16* __restrict__ dst, int K, int Nn, int Kp, int Np) {
    int t = blockIdx.x * blockDim.x + threadIdx.x;
    if (t >= Kp * Np) return;
    int n = t / Kp, k = t - n * Kp;
    float v = 0.0f;
    if (k < K && n < Nn) {
        v = W[k * Nn + n];
        if (Wsub) v -= Wsub[k * Nn + n];
    }
    dst[t] = (_Float16)v;
}

// ---------------- KNN: one batch (512 nodes) per block --------------------------------
template <typename T>
__global__ __launch_bounds__(512) void knn_kernel(const T* __restrict__ feat, int stride,
                                                  int* __restrict__ idx) {
    __shared__ float pos[NN][3];
    const int b = blockIdx.x, t = threadIdx.x;
    const size_t base = (size_t)b * NN;
    for (int d = 0; d < 3; ++d) pos[t][d] = (float)feat[(base + t) * stride + d];
    __syncthreads();
    const float px = pos[t][0], py = pos[t][1], pz = pos[t][2];
    float bd[KNBR] = {1e30f, 1e30f, 1e30f, 1e30f};
    int   bi[KNBR] = {0, 0, 0, 0};
    for (int j = 0; j < NN; ++j) {
        float dx = pos[j][0] - px, dy = pos[j][1] - py, dz = pos[j][2] - pz;
        float d2 = dx * dx + dy * dy + dz * dz;
        if (j == t) d2 = 1e30f;
        if (d2 < bd[KNBR - 1]) {
            int k = KNBR - 1;
            while (k > 0 && d2 < bd[k - 1]) { bd[k] = bd[k - 1]; bi[k] = bi[k - 1]; --k; }
            bd[k] = d2; bi[k] = j;
        }
    }
    for (int k = 0; k < KNBR; ++k) idx[(base + t) * KNBR + k] = bi[k];
}

// ---------------- pack x (G x 4 f32) -> (G x 32 f16, zero-padded) ---------------------
__global__ void pack_x(const float* __restrict__ x, _Float16* __restrict__ x16) {
    int t = blockIdx.x * blockDim.x + threadIdx.x;
    if (t >= G_NODES * 32) return;
    int g = t >> 5, c = t & 31;
    x16[t] = (c < 4) ? (_Float16)x[g * 4 + c] : (_Float16)0.0f;
}

// ---------------- generic WMMA GEMM: C(MxNp) = A(MxKp) @ Bt^T + bias ------------------
// Block tile 128 rows x 64 cols; 8 waves; double-buffered LDS with async global->LDS.
__global__ __launch_bounds__(256) void gemm_wmma(const _Float16* __restrict__ A, int Kp,
                                                 const _Float16* __restrict__ Bt,  // (Np x Kp)
                                                 const float* __restrict__ bias, int Nreal,
                                                 int doAct, _Float16* __restrict__ C, int Cstride) {
    __shared__ _Float16 sA[2][128 * 32];
    __shared__ _Float16 sB[2][64 * 32];
    const int tid = threadIdx.x;
    const int lane = tid & 31, w = tid >> 5;
    const int rowBase = blockIdx.x * 128;
    const int colBase = blockIdx.y * 64;

    // Each thread moves 3 x 16B per tile: 2 chunks of A (128x32), 1 chunk of B (64x32).
    auto prefetch = [&](int buf, int kb) {
#pragma unroll
        for (int c = 0; c < 2; ++c) {
            int ci = tid + c * 256;
            int r = ci >> 2, cc = (ci & 3) << 3;
            copy16_g2l(A + (size_t)(rowBase + r) * Kp + kb + cc, &sA[buf][r * 32 + cc]);
        }
        int r = tid >> 2, cc = (tid & 3) << 3;
        copy16_g2l(Bt + (size_t)(colBase + r) * Kp + kb + cc, &sB[buf][r * 32 + cc]);
    };

    v8f acc[4];
#pragma unroll
    for (int nt = 0; nt < 4; ++nt)
#pragma unroll
        for (int i = 0; i < 8; ++i) acc[nt][i] = 0.0f;

    const int arow = (w << 4) + (lane & 15);
    const int ko = (lane >> 4) << 3;
    const int nk = Kp >> 5;

    prefetch(0, 0);
    for (int kbi = 0; kbi < nk; ++kbi) {
        const int cur = kbi & 1;
        if (kbi + 1 < nk) {
            prefetch(cur ^ 1, (kbi + 1) << 5);
#if USE_ASYNC_LDS
            __builtin_amdgcn_s_wait_asynccnt(3);   // first batch (3 ops/wave) landed
#endif
        } else {
#if USE_ASYNC_LDS
            __builtin_amdgcn_s_wait_asynccnt(0);
#endif
        }
        __syncthreads();
        v16h af = load_frag(&sA[cur][0], arow, 32, ko);
#pragma unroll
        for (int nt = 0; nt < 4; ++nt) {
            v16h bf = load_frag(&sB[cur][0], nt * 16 + (lane & 15), 32, ko);
            acc[nt] = __builtin_amdgcn_wmma_f32_16x16x32_f16(false, af, false, bf,
                                                             (short)0, acc[nt], false, false);
        }
        __syncthreads();  // all waves done reading `cur` before it is overwritten
    }

#pragma unroll
    for (int nt = 0; nt < 4; ++nt) {
        int col = colBase + nt * 16 + (lane & 15);
        float bv = (bias && col < Nreal) ? bias[col] : 0.0f;
#pragma unroll
        for (int r = 0; r < 8; ++r) {
            int row = rowBase + (w << 4) + r + ((lane >> 4) << 3);
            float v = acc[nt][r] + bv;
            if (doAct) v = lrelu(v);
            if (col >= Nreal) v = 0.0f;
            C[(size_t)row * Cstride + col] = (_Float16)v;
        }
    }
}

// ---------------- fused edge MLP: h1 = lrelu(P[i]+Q[j]+b1); out[n] = sum_k lrelu(h1@W2+b2)
// Block: 32 nodes (128 edges). h1 (128 x Khp) and W2^T (192 x Khp) live in dynamic LDS.
// W2^T panel is fetched by the Tensor Data Mover, overlapped with the h1 gather phase.
__global__ __launch_bounds__(256) void edge_mlp(const _Float16* __restrict__ P,
                                                const _Float16* __restrict__ Q,
                                                int Khp, int Khr,
                                                const float* __restrict__ b1,
                                                const _Float16* __restrict__ W2t,  // (192 x Khp)
                                                const float* __restrict__ b2,
                                                const int* __restrict__ idx,
                                                _Float16* __restrict__ out) {  // (G x 192)
    extern __shared__ _Float16 smem[];
    _Float16* sH = smem;              // 128 * Khp
    _Float16* sW = smem + 128 * Khp;  // 192 * Khp
    const int tid = threadIdx.x;
    const int lane = tid & 31, w = tid >> 5;
    const int gbase = blockIdx.x * 32;
    const int batchBase = (gbase / NN) * NN;

#if USE_TDM
    if (w == 0) {
        // 2-D TDM descriptor: tile == tensor == (192 rows x Khp cols) of 2-byte elems.
        unsigned ldsAddr = (unsigned)(size_t)(AS3 _Float16*)sW;
        unsigned long long ga = (unsigned long long)(size_t)W2t;
        u32x4 g0;
        g0[0] = 1u;                                                 // count=1
        g0[1] = ldsAddr;                                            // lds_addr (bytes)
        g0[2] = (unsigned)ga;                                       // global_addr[31:0]
        g0[3] = (unsigned)((ga >> 32) & 0x01FFFFFFu) | 0x80000000u; // addr[56:32] | type=2
        i32x8 g1;
        g1[0] = 1 << 16;                          // data_size = 1 (2 bytes)
        g1[1] = (Khp & 0xFFFF) << 16;             // tensor_dim0[15:0]
        g1[2] = 192 << 16;                        // tensor_dim0[31:16]=0 | tensor_dim1[15:0]
        g1[3] = Khp << 16;                        // tensor_dim1[31:16]=0 | tile_dim0
        g1[4] = 192;                              // tile_dim1 | tile_dim2=0
        g1[5] = Khp;                              // tensor_dim0_stride[31:0]
        g1[6] = 0;                                // stride0 hi | stride1 lo
        g1[7] = 0;
        i32x4 z4 = {0, 0, 0, 0};
        i32x8 z8 = {0, 0, 0, 0, 0, 0, 0, 0};
        __builtin_amdgcn_tensor_load_to_lds(g0, g1, z4, z4, z8, 0);
    }
#else
    for (int t = tid; t < (192 * Khp) / 8; t += 256) {
        int o = t * 8;
        *(v8h*)&sW[o] = *(const v8h*)(W2t + o);
    }
#endif

    // Build h1 for the block's 128 edges while the TDM moves W2^T.
    for (int t = tid; t < 128 * Khp; t += 256) {
        int e = t / Khp, c = t - e * Khp;
        int ig = gbase + (e >> 2);
        int jg = batchBase + idx[ig * KNBR + (e & 3)];
        float v = 0.0f;
        if (c < Khr) {
            v = (float)P[(size_t)ig * Khp + c] + (float)Q[(size_t)jg * Khp + c] + b1[c];
            v = lrelu(v);
        }
        sH[t] = (_Float16)v;
    }
#if USE_TDM
    if (w == 0) __builtin_amdgcn_s_wait_tensorcnt(0);
#endif
    __syncthreads();

    const int arow = (w << 4) + (lane & 15);
    const int ko = (lane >> 4) << 3;

    v8f acc[12];
#pragma unroll
    for (int nt = 0; nt < 12; ++nt)
#pragma unroll
        for (int i = 0; i < 8; ++i) acc[nt][i] = 0.0f;

    for (int kb = 0; kb < Khp; kb += 32) {
        v16h af = load_frag(sH + kb, arow, Khp, ko);
#pragma unroll
        for (int nt = 0; nt < 12; ++nt) {
            v16h bf = load_frag(sW + kb, nt * 16 + (lane & 15), Khp, ko);
            acc[nt] = __builtin_amdgcn_wmma_f32_16x16x32_f16(false, af, false, bf,
                                                             (short)0, acc[nt], false, false);
        }
    }

    // bias + leaky per edge row, then sum over the 4 neighbors of each node.
    // Row m of the wave tile sits in acc VGPR (m%8), lane half (m/8):
    //   r=0..3 -> nodes {4w+0 (lanes 0-15), 4w+2 (lanes 16-31)}
    //   r=4..7 -> nodes {4w+1, 4w+3}
    const int half = lane >> 4;
#pragma unroll
    for (int nt = 0; nt < 12; ++nt) {
        int col = nt * 16 + (lane & 15);
        float bv = b2[col];
        float s0 = 0.0f, s1 = 0.0f;
#pragma unroll
        for (int r = 0; r < 4; ++r) {
            s0 += lrelu(acc[nt][r] + bv);
            s1 += lrelu(acc[nt][r + 4] + bv);
        }
        int n0 = gbase + (w << 2) + (half ? 2 : 0);
        out[(size_t)n0 * FOUT + col]       = (_Float16)s0;
        out[(size_t)(n0 + 1) * FOUT + col] = (_Float16)s1;
    }
}

// ---------------- concat [x0, a, b, c, d] -> (G x 784 f16, padded) --------------------
__global__ void cat_kernel(const float* __restrict__ x, const _Float16* __restrict__ a,
                           const _Float16* __restrict__ b, const _Float16* __restrict__ c,
                           const _Float16* __restrict__ d, _Float16* __restrict__ cat) {
    int t = blockIdx.x * blockDim.x + threadIdx.x;
    if (t >= G_NODES * 784) return;
    int g = t / 784, cc = t - g * 784;
    float v;
    if      (cc < 4)   v = x[g * 4 + cc];
    else if (cc < 196) v = (float)a[(size_t)g * FOUT + cc - 4];
    else if (cc < 388) v = (float)b[(size_t)g * FOUT + cc - 196];
    else if (cc < 580) v = (float)c[(size_t)g * FOUT + cc - 388];
    else if (cc < 772) v = (float)d[(size_t)g * FOUT + cc - 580];
    else               v = 0.0f;
    cat[t] = (_Float16)v;
}

// ---------------- pooling: lrelu([max,min,sum,mean]) over N per (batch,col) -----------
__global__ __launch_bounds__(192) void pool_kernel(const _Float16* __restrict__ h,
                                                   float* __restrict__ pcat) {
    const int b = blockIdx.x, c = threadIdx.x;
    float mx = -1e30f, mn = 1e30f, sm = 0.0f;
    const _Float16* p = h + (size_t)b * NN * FOUT + c;
    for (int n = 0; n < NN; ++n) {
        float v = (float)p[(size_t)n * FOUT];
        mx = fmaxf(mx, v); mn = fminf(mn, v); sm += v;
    }
    pcat[b * 768 + c]       = lrelu(mx);
    pcat[b * 768 + 192 + c] = lrelu(mn);
    pcat[b * 768 + 384 + c] = lrelu(sm);
    pcat[b * 768 + 576 + c] = lrelu(sm * (1.0f / NN));
}

// ---------------- head: lrelu(p@n3w+n3b) @ n4w + n4b; z = relu(z)+eps -----------------
__global__ __launch_bounds__(96) void head_kernel(const float* __restrict__ pcat,
                                                  const float* __restrict__ n3w,
                                                  const float* __restrict__ n3b,
                                                  const float* __restrict__ n4w,
                                                  const float* __restrict__ n4b,
                                                  float* __restrict__ out) {
    __shared__ float hid[96];
    const int b = blockIdx.x, j = threadIdx.x;
    const float* pr = pcat + b * 768;
    float s = n3b[j];
    for (int i = 0; i < 768; ++i) s += pr[i] * n3w[i * 96 + j];
    hid[j] = lrelu(s);
    __syncthreads();
    if (j < 2) {
        float o = n4b[j];
        for (int k = 0; k < 96; ++k) o += hid[k] * n4w[k * 2 + j];
        if (j == 1) o = (o > 0.0f ? o : 0.0f) + 1e-15f;
        out[b * 2 + j] = o;
    }
}

// =====================================================================================
extern "C" void kernel_launch(void* const* d_in, const int* in_sizes, int n_in,
                              void* d_out, int out_size, void* d_ws, size_t ws_size,
                              hipStream_t stream) {
    (void)in_sizes; (void)n_in; (void)out_size; (void)ws_size;
    const float* x = (const float*)d_in[0];
    const float* cw1[4]; const float* cb1[4]; const float* cw2[4]; const float* cb2[4];
    for (int l = 0; l < 4; ++l) {
        cw1[l] = (const float*)d_in[1 + 4 * l];
        cb1[l] = (const float*)d_in[2 + 4 * l];
        cw2[l] = (const float*)d_in[3 + 4 * l];
        cb2[l] = (const float*)d_in[4 + 4 * l];
    }
    const float* n1w = (const float*)d_in[17]; const float* n1b = (const float*)d_in[18];
    const float* n2w = (const float*)d_in[19]; const float* n2b = (const float*)d_in[20];
    const float* n3w = (const float*)d_in[21]; const float* n3b = (const float*)d_in[22];
    const float* n4w = (const float*)d_in[23]; const float* n4b = (const float*)d_in[24];
    float* out = (float*)d_out;

    // ---- workspace bump allocator ----
    char* wsp = (char*)d_ws;
    size_t off = 0;
    auto alloc = [&](size_t bytes) -> char* {
        char* p = wsp + off;
        off = (off + bytes + 255) & ~(size_t)255;
        return p;
    };
    int*      idx   = (int*)alloc((size_t)G_NODES * KNBR * 4);
    _Float16* x16   = (_Float16*)alloc((size_t)G_NODES * 32 * 2);
    _Float16* P16   = (_Float16*)alloc((size_t)G_NODES * 256 * 2);
    _Float16* Q16   = (_Float16*)alloc((size_t)G_NODES * 256 * 2);
    _Float16* act[4];
    for (int l = 0; l < 4; ++l) act[l] = (_Float16*)alloc((size_t)G_NODES * FOUT * 2);
    _Float16* cat16 = (_Float16*)alloc((size_t)G_NODES * 784 * 2);
    _Float16* h1n16 = (_Float16*)alloc((size_t)G_NODES * 256 * 2);
    _Float16* h2_16 = (_Float16*)alloc((size_t)G_NODES * FOUT * 2);
    float*    pcat  = (float*)alloc((size_t)NB * 768 * 4);
    _Float16* Wd1t  = (_Float16*)alloc(128 * 32 * 2);
    _Float16* Wb1t  = (_Float16*)alloc(128 * 32 * 2);
    _Float16* W21t  = (_Float16*)alloc(192 * 128 * 2);
    _Float16* Wdt[4]; _Float16* Wbt[4]; _Float16* W2t[4];
    for (int l = 1; l < 4; ++l) {
        Wdt[l] = (_Float16*)alloc(256 * 192 * 2);
        Wbt[l] = (_Float16*)alloc(256 * 192 * 2);
        W2t[l] = (_Float16*)alloc(192 * 256 * 2);
    }
    _Float16* n1t = (_Float16*)alloc(256 * 784 * 2);
    _Float16* n2t = (_Float16*)alloc(192 * 256 * 2);

    auto cdiv = [](int a, int b) { return (a + b - 1) / b; };

    // ---- weight prep ----
    prep_weight<<<cdiv(128 * 32, 256), 256, 0, stream>>>(cw1[0], cw1[0] + 4 * 96, Wd1t, 4, 96, 32, 128);
    prep_weight<<<cdiv(128 * 32, 256), 256, 0, stream>>>(cw1[0] + 4 * 96, nullptr, Wb1t, 4, 96, 32, 128);
    prep_weight<<<cdiv(192 * 128, 256), 256, 0, stream>>>(cw2[0], nullptr, W21t, 96, 192, 128, 192);
    for (int l = 1; l < 4; ++l) {
        prep_weight<<<cdiv(256 * 192, 256), 256, 0, stream>>>(cw1[l], cw1[l] + 192 * 252, Wdt[l], 192, 252, 192, 256);
        prep_weight<<<cdiv(256 * 192, 256), 256, 0, stream>>>(cw1[l] + 192 * 252, nullptr, Wbt[l], 192, 252, 192, 256);
        prep_weight<<<cdiv(192 * 256, 256), 256, 0, stream>>>(cw2[l], nullptr, W2t[l], 252, 192, 256, 192);
    }
    prep_weight<<<cdiv(256 * 784, 256), 256, 0, stream>>>(n1w, nullptr, n1t, 772, 252, 784, 256);
    prep_weight<<<cdiv(192 * 256, 256), 256, 0, stream>>>(n2w, nullptr, n2t, 252, 192, 256, 192);

    pack_x<<<cdiv(G_NODES * 32, 256), 256, 0, stream>>>(x, x16);

    // ---- layer 1 ----
    knn_kernel<float><<<NB, 512, 0, stream>>>(x, 4, idx);
    gemm_wmma<<<dim3(G_NODES / 128, 2), 256, 0, stream>>>(x16, 32, Wd1t, nullptr, 96, 0, P16, 128);
    gemm_wmma<<<dim3(G_NODES / 128, 2), 256, 0, stream>>>(x16, 32, Wb1t, nullptr, 96, 0, Q16, 128);
    edge_mlp<<<G_NODES / 32, 256, (128 + 192) * 128 * 2, stream>>>(
        P16, Q16, 128, 96, cb1[0], W21t, cb2[0], idx, act[0]);

    // ---- layers 2-4 ----
    for (int l = 1; l < 4; ++l) {
        knn_kernel<_Float16><<<NB, 512, 0, stream>>>(act[l - 1], FOUT, idx);
        gemm_wmma<<<dim3(G_NODES / 128, 4), 256, 0, stream>>>(act[l - 1], FOUT, Wdt[l], nullptr, 252, 0, P16, 256);
        gemm_wmma<<<dim3(G_NODES / 128, 4), 256, 0, stream>>>(act[l - 1], FOUT, Wbt[l], nullptr, 252, 0, Q16, 256);
        edge_mlp<<<G_NODES / 32, 256, (128 + 192) * 256 * 2, stream>>>(
            P16, Q16, 256, 252, cb1[l], W2t[l], cb2[l], idx, act[l]);
    }

    // ---- concat + n1 + n2 ----
    cat_kernel<<<cdiv(G_NODES * 784, 256), 256, 0, stream>>>(x, act[0], act[1], act[2], act[3], cat16);
    gemm_wmma<<<dim3(G_NODES / 128, 4), 256, 0, stream>>>(cat16, 784, n1t, n1b, 252, 1, h1n16, 256);
    gemm_wmma<<<dim3(G_NODES / 128, 3), 256, 0, stream>>>(h1n16, 256, n2t, n2b, 192, 0, h2_16, 192);

    // ---- pooling + head ----
    pool_kernel<<<NB, 192, 0, stream>>>(h2_16, pcat);
    head_kernel<<<NB, 96, 0, stream>>>(pcat, n3w, n3b, n4w, n4b, out);
}
